// GATDiffPooled_86225763435137
// MI455X (gfx1250) — matrix operand order repaired and checked by reading, not compiled, not run on hardware
//
#include <hip/hip_runtime.h>
#include <hip/hip_bf16.h>
#include <math.h>

// ---------------------------------------------------------------------------
// GATDiffPooled for MI455X (gfx1250, wave32, WMMA).
// All dense projections run through v_wmma_f32_16x16x32_bf16 (bf16 in, f32
// accumulate).  Sparse attention uses atomic segment-softmax.  Temporal GAT
// layers reduce exactly to a shifted copy (1 in-edge per dst => att==1).
// ---------------------------------------------------------------------------

#define T_WIN   28
#define NN      512
#define FIN     26
#define HIDD    256
#define NHEAD   4
#define EDG     4096
#define KK1     14
#define KK2     5

typedef __bf16 v16bf __attribute__((ext_vector_type(16)));
typedef __bf16 v8bf  __attribute__((ext_vector_type(8)));
typedef __bf16 v2bf  __attribute__((ext_vector_type(2)));
typedef float  v8f   __attribute__((ext_vector_type(8)));
typedef float  v2f   __attribute__((ext_vector_type(2)));

// ------------------------- workspace layout (floats) -----------------------
#define B_H1    ((size_t)0)                       // 14336*308 concat input
#define B_PROJ  (B_H1   + (size_t)14336*308)      // 14336*1024 projection
#define B_OUT   (B_PROJ + (size_t)14336*1024)     // 14336*1024 gat out / ping
#define B_ACT1  (B_OUT  + (size_t)14336*1024)     // 14336*256
#define B_ACT2  (B_ACT1 + (size_t)14336*256)      // 14336*256
#define B_EW    (B_ACT2 + (size_t)14336*256)      // 4096*8
#define B_MT1   (B_EW   + (size_t)4096*8)         // 4096*128
#define B_MT2   (B_MT1  + (size_t)4096*128)       // 4096*128
#define B_EPROJ (B_MT2  + (size_t)4096*128)       // 4096*1024
#define B_EDOT  (B_EPROJ+ (size_t)4096*1024)      // 4096*4
#define B_HSD   (B_EDOT + (size_t)4096*4)         // 14336*4
#define B_HDD   (B_HSD  + (size_t)14336*4)        // 14336*4
#define B_MX    (B_HDD  + (size_t)14336*4)        // 14336*4
#define B_DEN   (B_MX   + (size_t)14336*4)        // 14336*4
#define B_EX    (B_DEN  + (size_t)14336*4)        // 28*4096*4
#define B_S     (B_EX   + (size_t)28*4096*4)      // 14336*14 pool logits
#define B_XP    (B_S    + (size_t)14336*14)       // 14*512*256
#define B_AP    (B_XP   + (size_t)14*512*256)     // 512*14*14
#define B_TMP   (B_AP   + (size_t)512*14*14)      // 14336*26 fourier tmp
#define B_X2    (B_TMP  + (size_t)14336*26)       // 5*512*256
#define B_MU    (B_X2   + (size_t)5*512*256)      // 512
#define B_VAR   (B_MU   + (size_t)512)            // 512

// ---------------------------------------------------------------------------
// GEMM: C[M,N] = act(A[M,K] @ B[K,N] + bias), bf16 WMMA, f32 accumulate.
// One wave computes a 16x32 tile (two 16x16 WMMA accumulators sharing the A
// fragment).  K consumed 32 at a time through LDS:
//   As  [16][32]  row-major   -> A fragment = two contiguous ds_load_b128
//   Bst [32][32]  TRANSPOSED  -> B fragment = two contiguous ds_load_b128
// Fully-in-bounds tiles (the dominant GEMMs here) take a uniform fast path:
// float2 global loads, packed bf16 LDS stores, zero guards.  Ragged tiles
// take a branchless clamped path.
// ---------------------------------------------------------------------------
__global__ void gemm_bf16_kernel(const float* __restrict__ A, int lda,
                                 const float* __restrict__ B, int ldb,
                                 const float* __restrict__ bias,
                                 float* __restrict__ C, int ldc,
                                 int M, int K, int N, int act)
{
    __shared__ __align__(16) __bf16 As[16][32];
    __shared__ __align__(16) __bf16 Bst[32][32];   // [col][k]
    const int row0 = blockIdx.x * 16;
    const int col0 = blockIdx.y * 32;
    const int tid  = threadIdx.x;                  // 0..31 (wave32)

    v8f acc0 = {}, acc1 = {};
    const int rA  = tid & 15;
    const int kbA = (tid >> 4) * 8;                // A frag: lanes16-31 at K=8
    const int kbB = (tid >> 4) * 8;                // B frag: lanes16-31 at K=8
    const bool fullMN = (row0 + 16 <= M) && (col0 + 32 <= N) && (((lda | ldb) & 1) == 0);

    for (int kb = 0; kb < K; kb += 32) {
        if (kb + 32 < K) {                          // prefetch next K-slab
            __builtin_prefetch(&A[(size_t)min(row0 + rA, M - 1) * lda + kb + 32], 0, 1);
            __builtin_prefetch(&B[(size_t)(kb + 32 + (tid & 15)) * ldb + col0], 0, 1);
        }
        if (fullMN && (kb + 32 <= K)) {
            // ---------- uniform fast path: no guards, float2 loads ----------
#pragma unroll
            for (int i = 0; i < 8; ++i) {           // A tile 16x32
                int r   = i * 2 + (tid >> 4);
                int kk2 = (tid & 15) * 2;
                v2f av = *(const v2f*)&A[(size_t)(row0 + r) * lda + kb + kk2];
                v2bf pk; pk[0] = (__bf16)av[0]; pk[1] = (__bf16)av[1];
                *(v2bf*)&As[r][kk2] = pk;
            }
#pragma unroll
            for (int i = 0; i < 16; ++i) {          // B tile 32k x 32n (transposed store)
                int k  = i * 2 + (tid >> 4);
                int c2 = (tid & 15) * 2;
                v2f bv = *(const v2f*)&B[(size_t)(kb + k) * ldb + col0 + c2];
                Bst[c2][k]     = (__bf16)bv[0];
                Bst[c2 + 1][k] = (__bf16)bv[1];
            }
        } else {
            // ---------- ragged path: branchless clamped loads ----------
#pragma unroll
            for (int i = 0; i < 16; ++i) {
                int idx = i * 32 + tid;
                int r = idx >> 5, kk = idx & 31;
                int gr = row0 + r, gk = kb + kk;
                int cr = gr < M ? gr : M - 1;
                int ck = gk < K ? gk : K - 1;
                float av = A[(size_t)cr * lda + ck];
                av = (gr < M && gk < K) ? av : 0.f;
                As[r][kk] = (__bf16)av;
            }
#pragma unroll
            for (int i = 0; i < 32; ++i) {
                int idx = i * 32 + tid;
                int k = idx >> 5, c = idx & 31;
                int gk = kb + k, gc = col0 + c;
                int ckk = gk < K ? gk : K - 1;
                int ccc = gc < N ? gc : N - 1;
                float bv = B[(size_t)ckk * ldb + ccc];
                bv = (gk < K && gc < N) ? bv : 0.f;
                Bst[c][k] = (__bf16)bv;
            }
        }
        __syncthreads();

        // ---- fragments: 6 x ds_load_b128 ----
        v8bf a0 = *(const v8bf*)&As[rA][kbA];
        v8bf a1 = *(const v8bf*)&As[rA][kbA + 16];
        v16bf af = __builtin_shufflevector(a0, a1, 0,1,2,3,4,5,6,7,8,9,10,11,12,13,14,15);
        v8bf b0l = *(const v8bf*)&Bst[rA][kbB];
        v8bf b0h = *(const v8bf*)&Bst[rA][kbB + 16];
        v16bf bf0 = __builtin_shufflevector(b0l, b0h, 0,1,2,3,4,5,6,7,8,9,10,11,12,13,14,15);
        v8bf b1l = *(const v8bf*)&Bst[rA + 16][kbB];
        v8bf b1h = *(const v8bf*)&Bst[rA + 16][kbB + 16];
        v16bf bf1 = __builtin_shufflevector(b1l, b1h, 0,1,2,3,4,5,6,7,8,9,10,11,12,13,14,15);

        acc0 = __builtin_amdgcn_wmma_f32_16x16x32_bf16(false, af, false, bf0, (short)0, acc0, false, false);
        acc1 = __builtin_amdgcn_wmma_f32_16x16x32_bf16(false, af, false, bf1, (short)0, acc1, false, false);
        __syncthreads();
    }

    const int col = tid & 15;
    const int rb  = (tid >> 4) * 8;
#pragma unroll
    for (int r = 0; r < 8; ++r) {
        int gr = row0 + rb + r;
        if (gr >= M) continue;
        int gc0 = col0 + col, gc1 = col0 + 16 + col;
        if (gc0 < N) {
            float v = acc0[r];
            if (bias) v += bias[gc0];
            if (act == 1) v = v > 0.f ? v : expm1f(v);
            C[(size_t)gr * ldc + gc0] = v;
        }
        if (gc1 < N) {
            float v = acc1[r];
            if (bias) v += bias[gc1];
            if (act == 1) v = v > 0.f ? v : expm1f(v);
            C[(size_t)gr * ldc + gc1] = v;
        }
    }
}

// ------------------------------ small kernels ------------------------------
__global__ void fill_kernel(float* p, float v, size_t n) {
    size_t i = (size_t)blockIdx.x * blockDim.x + threadIdx.x;
    if (i < n) p[i] = v;
}

__global__ void bias_init_kernel(float* out, const float* bias, size_t M, int F) {
    size_t i = (size_t)blockIdx.x * blockDim.x + threadIdx.x;
    if (i >= M * (size_t)F) return;
    out[i] = bias[i % F];
}

// mode 0: dst=elu(a); mode 1: dst=a+elu(b); mode 2: dst=a+b
__global__ void ew_kernel(float* dst, const float* a, const float* b, size_t n, int mode) {
    size_t i = (size_t)blockIdx.x * blockDim.x + threadIdx.x;
    if (i >= n) return;
    float v;
    if (mode == 0)      { float x = a[i]; v = x > 0.f ? x : expm1f(x); }
    else if (mode == 1) { float x = b[i]; v = a[i] + (x > 0.f ? x : expm1f(x)); }
    else                { v = a[i] + b[i]; }
    dst[i] = v;
}

__global__ void fourier_kernel(const float* tmp, float* h1, int M) {
    int i = blockIdx.x * blockDim.x + threadIdx.x;
    if (i >= M * FIN) return;
    int m = i / FIN, j = i % FIN;
    float xp = 6.283185307179586f * tmp[i];
    h1[(size_t)m * (HIDD + 2 * FIN) + HIDD + j]       = sinf(xp);
    h1[(size_t)m * (HIDD + 2 * FIN) + HIDD + FIN + j] = cosf(xp);
}

// d[m,h] = sum_c X[m, h*C+c] * a[h,c]
__global__ void dot_heads_kernel(const float* X, const float* avec, float* out,
                                 int M, int H, int C) {
    int i = blockIdx.x * blockDim.x + threadIdx.x;
    if (i >= M * H) return;
    int m = i / H, h = i % H;
    const float* xp = X + (size_t)m * H * C + (size_t)h * C;
    const float* ap = avec + (size_t)h * C;
    float s = 0.f;
    for (int c = 0; c < C; ++c) s += xp[c] * ap[c];
    out[i] = s;
}

__device__ inline void atomicMaxF(float* addr, float v) {
    unsigned int* ua = (unsigned int*)addr;
    unsigned int old = *ua;
    while (true) {
        float f = __uint_as_float(old);
        if (f >= v) break;
        unsigned int assumed = old;
        old = atomicCAS(ua, assumed, __float_as_uint(v));
        if (old == assumed) break;
    }
}

__global__ void gat_logits_kernel(const float* hsd, const float* hdd, const float* edot,
                                  const int* src, const int* dst,
                                  float* al, float* mx, int T, int N, int E, int H) {
    int i = blockIdx.x * blockDim.x + threadIdx.x;
    if (i >= T * E * H) return;
    int h = i % H, e = (i / H) % E, t = i / (H * E);
    int s = src[e], d = dst[e];
    float v = hsd[(size_t)(t * N + s) * H + h] + hdd[(size_t)(t * N + d) * H + h]
            + edot[(size_t)e * H + h];
    v = v > 0.f ? v : 0.2f * v;                 // leaky_relu
    al[i] = v;
    atomicMaxF(&mx[(size_t)(t * N + d) * H + h], v);
}

__global__ void gat_exp_den_kernel(float* al, const float* mx, float* den,
                                   const int* dst, int T, int N, int E, int H) {
    int i = blockIdx.x * blockDim.x + threadIdx.x;
    if (i >= T * E * H) return;
    int h = i % H, e = (i / H) % E, t = i / (H * E);
    int d = dst[e];
    float ex = expf(al[i] - mx[(size_t)(t * N + d) * H + h]);
    al[i] = ex;
    atomicAdd(&den[(size_t)(t * N + d) * H + h], ex);
}

__global__ void gat_scatter_kernel(const float* proj, const float* ex, const float* den,
                                   const int* src, const int* dst, float* out,
                                   int T, int N, int E, int H, int C) {
    int blk = blockIdx.x;                 // t*E + e
    int e = blk % E, t = blk / E;
    int s = src[e], d = dst[e];
    int HC = H * C;
    const float* ps  = proj + (size_t)(t * N + s) * HC;
    float*       po  = out  + (size_t)(t * N + d) * HC;
    const float* exb = ex   + (size_t)blk * H;
    const float* deb = den  + (size_t)(t * N + d) * H;
    for (int j = threadIdx.x; j < HC; j += blockDim.x) {
        float att = exb[j / C] / deb[j / C];
        atomicAdd(&po[j], att * ps[j]);
    }
}

// temporal GAT: each dst has exactly one in-edge -> att==1 exactly.
__global__ void temporal_kernel(const float* proj, const float* bias, float* out,
                                int N, int F, int total) {
    int i = blockIdx.x * blockDim.x + threadIdx.x;
    if (i >= total) return;
    int f = i % F, m = i / F;
    int srcm = (m < N) ? m : m - N;
    float v = proj[(size_t)srcm * F + f] + bias[f];
    out[i] = v > 0.f ? v : expm1f(v);
}

__global__ void bn_feat_stats_kernel(const float* a, const float* b,
                                     float* mu, float* var, int M, int F) {
    int f = blockIdx.x;
    __shared__ float ss[256], ss2[256];
    float s = 0.f, s2 = 0.f;
    for (int m = threadIdx.x; m < M; m += blockDim.x) {
        float v = a[(size_t)m * F + f];
        if (b) v += b[(size_t)m * F + f];
        s += v; s2 += v * v;
    }
    ss[threadIdx.x] = s; ss2[threadIdx.x] = s2;
    __syncthreads();
    for (int o = blockDim.x >> 1; o > 0; o >>= 1) {
        if (threadIdx.x < (unsigned)o) { ss[threadIdx.x] += ss[threadIdx.x + o]; ss2[threadIdx.x] += ss2[threadIdx.x + o]; }
        __syncthreads();
    }
    if (threadIdx.x == 0) {
        float m1 = ss[0] / M;
        mu[f] = m1; var[f] = ss2[0] / M - m1 * m1;
    }
}

__global__ void bn_feat_apply_kernel(const float* a, const float* b, float* out,
                                     const float* mu, const float* var,
                                     const float* g, const float* bb, int M, int F) {
    size_t i = (size_t)blockIdx.x * blockDim.x + threadIdx.x;
    if (i >= (size_t)M * F) return;
    int f = i % F;
    float v = a[i]; if (b) v += b[i];
    out[i] = (v - mu[f]) * rsqrtf(var[f] + 1e-5f) * g[f] + bb[f];
}

__global__ void bn_nodes_stats_kernel(const float* x, float* mu, float* var,
                                      int T, int N, int F) {
    int n = blockIdx.x;
    __shared__ float ss[256], ss2[256];
    float s = 0.f, s2 = 0.f;
    int tot = T * F;
    for (int i = threadIdx.x; i < tot; i += blockDim.x) {
        float v = x[((size_t)(i / F) * N + n) * F + (i % F)];
        s += v; s2 += v * v;
    }
    ss[threadIdx.x] = s; ss2[threadIdx.x] = s2;
    __syncthreads();
    for (int o = blockDim.x >> 1; o > 0; o >>= 1) {
        if (threadIdx.x < (unsigned)o) { ss[threadIdx.x] += ss[threadIdx.x + o]; ss2[threadIdx.x] += ss2[threadIdx.x + o]; }
        __syncthreads();
    }
    if (threadIdx.x == 0) { float m1 = ss[0] / tot; mu[n] = m1; var[n] = ss2[0] / tot - m1 * m1; }
}

__global__ void bn_nodes_apply_kernel(float* x, const float* mu, const float* var,
                                      const float* g, const float* b, int T, int N, int F) {
    size_t i = (size_t)blockIdx.x * blockDim.x + threadIdx.x;
    if (i >= (size_t)T * N * F) return;
    int n = (int)((i / F) % N);
    x[i] = (x[i] - mu[n]) * rsqrtf(var[n] + 1e-5f) * g[n] + b[n];
}

__global__ void softmax_rows_kernel(float* x, int M, int K) {
    int m = blockIdx.x * blockDim.x + threadIdx.x;
    if (m >= M) return;
    float* r = x + (size_t)m * K;
    float mx = r[0];
    for (int k = 1; k < K; ++k) mx = fmaxf(mx, r[k]);
    float s = 0.f;
    for (int k = 0; k < K; ++k) { float e = expf(r[k] - mx); r[k] = e; s += e; }
    float inv = 1.f / s;
    for (int k = 0; k < K; ++k) r[k] *= inv;
}

// xp[k,n,f] = sum_t s[t,n,k] * x[t,n,f]
__global__ void pool_x_kernel(const float* s, const float* x, float* xp,
                              int T, int N, int F, int K) {
    size_t i = (size_t)blockIdx.x * blockDim.x + threadIdx.x;
    if (i >= (size_t)K * N * F) return;
    int f = i % F, n = (int)((i / F) % N), k = (int)(i / ((size_t)F * N));
    float acc = 0.f;
    for (int t = 0; t < T; ++t)
        acc += s[((size_t)t * N + n) * K + k] * x[((size_t)t * N + n) * F + f];
    xp[i] = acc;
}

// Ap[n,k,l] = s[0,n,k]*s[0,n,l] + sum_{t<T-1} s[t,n,k]*s[t+1,n,l]   (chain A)
__global__ void pool_A_kernel(const float* s, float* Ap, int T, int N, int K) {
    int i = blockIdx.x * blockDim.x + threadIdx.x;
    if (i >= N * K * K) return;
    int l = i % K, k = (i / K) % K, n = i / (K * K);
    float acc = s[((size_t)0 * N + n) * K + k] * s[((size_t)0 * N + n) * K + l];
    for (int t = 0; t < T - 1; ++t)
        acc += s[((size_t)t * N + n) * K + k] * s[((size_t)(t + 1) * N + n) * K + l];
    Ap[i] = acc;
}

// tg2: dense per-(n,l) attention over K1 in-edges, De=1 edge attr = Ap.
__global__ void tg2_kernel(const float* proj, const float* hsd, const float* hdd,
                           const float* Ap, const float* We, const float* aedge,
                           const float* bias, float* out, int N, int K, int H, int C) {
    int n = blockIdx.x / K, l = blockIdx.x % K;
    __shared__ float ce[8];
    __shared__ float att[16][8];
    int tid = threadIdx.x;          // 64
    int HC = H * C;
    if (tid < H) {
        float s = 0.f;
        for (int c = 0; c < C; ++c) s += We[tid * C + c] * aedge[tid * C + c];
        ce[tid] = s;
    }
    __syncthreads();
    if (tid < K * H) {
        int k = tid / H, h = tid % H;
        float w = Ap[((size_t)n * K + k) * K + l];
        float v = hsd[((size_t)k * N + n) * H + h] + hdd[((size_t)l * N + n) * H + h] + w * ce[h];
        att[k][h] = v > 0.f ? v : 0.2f * v;
    }
    __syncthreads();
    if (tid < H) {
        float m = -3.0e38f;
        for (int k = 0; k < K; ++k) m = fmaxf(m, att[k][tid]);
        float s = 0.f;
        for (int k = 0; k < K; ++k) { float e = expf(att[k][tid] - m); att[k][tid] = e; s += e; }
        float inv = 1.f / s;
        for (int k = 0; k < K; ++k) att[k][tid] *= inv;
    }
    __syncthreads();
    for (int j = tid; j < HC; j += blockDim.x) {
        int h = j / C;
        float acc = 0.f;
        for (int k = 0; k < K; ++k) acc += att[k][h] * proj[((size_t)k * N + n) * HC + j];
        out[((size_t)l * N + n) * HC + j] = acc + bias[j];
    }
}

// ------------------------------- host side ---------------------------------
static inline unsigned cdiv(size_t a, unsigned b) { return (unsigned)((a + b - 1) / b); }

static inline void g_gemm(const float* A, int lda, const float* B, int ldb,
                          const float* bias, float* C, int ldc,
                          int M, int K, int N, int act, hipStream_t s) {
    dim3 grid((M + 15) / 16, (N + 31) / 32);
    gemm_bf16_kernel<<<grid, 32, 0, s>>>(A, lda, B, ldb, bias, C, ldc, M, K, N, act);
}

struct GatP { const float *W, *We, *asrc, *adst, *aedge, *b; };

extern "C" void kernel_launch(void* const* d_in, const int* in_sizes, int n_in,
                              void* d_out, int out_size, void* d_ws, size_t ws_size,
                              hipStream_t stream) {
    (void)in_sizes; (void)n_in; (void)out_size; (void)ws_size;
    int ix = 0;
    auto F = [&](int i) { return (const float*)d_in[i]; };

    const float* x           = F(ix++);                    // [28,512,26]
    const int*   edge_index  = (const int*)d_in[ix++];     // [2,4096]
    const float* edge_weight = F(ix++);                    // [4096]
    const int* e_src = edge_index, *e_dst = edge_index + EDG;

    const float* fourier_b = F(ix++);
    const float* enc_W = F(ix++); const float* enc_b = F(ix++);
    const float *mW[5][3], *mB[5][3];                      // emlp, emlp0..3
    for (int l = 0; l < 5; ++l)
        for (int j = 0; j < 3; ++j) { mW[l][j] = F(ix++); mB[l][j] = F(ix++); }
    auto loadGat = [&]() { GatP g; g.W = F(ix++); g.We = F(ix++); g.asrc = F(ix++);
                           g.adst = F(ix++); g.aedge = F(ix++); g.b = F(ix++); return g; };
    GatP sg   = loadGat(), sg0 = loadGat(), sg1 = loadGat();
    GatP tg1[4]; for (int i = 0; i < 4; ++i) tg1[i] = loadGat();
    GatP sg2g = loadGat(), tg2g = loadGat(), goutp = loadGat();
    const float* sn_g = F(ix++); const float* sn_b = F(ix++);
    const float* n1_g = F(ix++); const float* n1_b = F(ix++);
    const float* n2_g = F(ix++); const float* n2_b = F(ix++);
    const float* pool1_W = F(ix++); const float* pool2_W = F(ix++);

    float* ws    = (float*)d_ws;
    float* bH1   = ws + B_H1;   float* bPROJ = ws + B_PROJ; float* bOUT = ws + B_OUT;
    float* bACT1 = ws + B_ACT1; float* bACT2 = ws + B_ACT2; float* bEW  = ws + B_EW;
    float* bMT1  = ws + B_MT1;  float* bMT2  = ws + B_MT2;  float* bEPR = ws + B_EPROJ;
    float* bEDOT = ws + B_EDOT; float* bHSD  = ws + B_HSD;  float* bHDD = ws + B_HDD;
    float* bMX   = ws + B_MX;   float* bDEN  = ws + B_DEN;  float* bEXV = ws + B_EX;
    float* bS    = ws + B_S;    float* bXP   = ws + B_XP;   float* bAP  = ws + B_AP;
    float* bTMP  = ws + B_TMP;  float* bX2   = ws + B_X2;
    float* bMU   = ws + B_MU;   float* bVAR  = ws + B_VAR;

    const int M = T_WIN * NN;        // 14336

    // generic 3-pass spatial GAT (shared edge list across T graphs)
    auto spatial_gat = [&](const float* proj, const GatP& p, int T, int H, int C, float* out) {
        int HC = H * C, Mn = T * NN;
        dot_heads_kernel<<<cdiv((size_t)Mn * H, 256), 256, 0, stream>>>(proj, p.asrc, bHSD, Mn, H, C);
        dot_heads_kernel<<<cdiv((size_t)Mn * H, 256), 256, 0, stream>>>(proj, p.adst, bHDD, Mn, H, C);
        dot_heads_kernel<<<cdiv((size_t)EDG * H, 256), 256, 0, stream>>>(bEPR, p.aedge, bEDOT, EDG, H, C);
        fill_kernel<<<cdiv((size_t)Mn * H, 256), 256, 0, stream>>>(bMX, -3.0e38f, (size_t)Mn * H);
        fill_kernel<<<cdiv((size_t)Mn * H, 256), 256, 0, stream>>>(bDEN, 0.f, (size_t)Mn * H);
        gat_logits_kernel<<<cdiv((size_t)T * EDG * H, 256), 256, 0, stream>>>(
            bHSD, bHDD, bEDOT, e_src, e_dst, bEXV, bMX, T, NN, EDG, H);
        gat_exp_den_kernel<<<cdiv((size_t)T * EDG * H, 256), 256, 0, stream>>>(
            bEXV, bMX, bDEN, e_dst, T, NN, EDG, H);
        bias_init_kernel<<<cdiv((size_t)Mn * HC, 256), 256, 0, stream>>>(out, p.b, (size_t)Mn, HC);
        gat_scatter_kernel<<<T * EDG, 256, 0, stream>>>(proj, bEXV, bDEN, e_src, e_dst,
                                                        out, T, NN, EDG, H, C);
    };
    auto run_mlp3 = [&](const float* in, int d0, int lidx, int d1, int d2) {
        g_gemm(in,  d0,  mW[lidx][0], d1, mB[lidx][0], bMT1, d1, EDG, d0, d1, 1, stream);
        g_gemm(bMT1, d1, mW[lidx][1], d2, mB[lidx][1], bMT2, d2, EDG, d1, d2, 1, stream);
        g_gemm(bMT2, d2, mW[lidx][2], 8,  mB[lidx][2], bEW,  8,  EDG, d2, 8,  0, stream);
    };

    // ---- encoder + fourier -> h1 [14336, 308] ----
    g_gemm(x, FIN, enc_W, HIDD, enc_b, bH1, HIDD + 2 * FIN, M, FIN, HIDD, 0, stream);
    g_gemm(x, FIN, fourier_b, FIN, nullptr, bTMP, FIN, M, FIN, FIN, 0, stream);
    fourier_kernel<<<cdiv((size_t)M * FIN, 256), 256, 0, stream>>>(bTMP, bH1, M);

    // ---- edge MLP: ew = mlp(edge_weight[:,None]) ----
    run_mlp3(edge_weight, 1, 0, 128, 128);

    // ---- sg: Fin=308, H=4, C=256 ----
    g_gemm(bH1, HIDD + 2 * FIN, sg.W, NHEAD * HIDD, nullptr, bPROJ, NHEAD * HIDD,
           M, HIDD + 2 * FIN, NHEAD * HIDD, 0, stream);
    g_gemm(bEW, 8, sg.We, NHEAD * HIDD, nullptr, bEPR, NHEAD * HIDD, EDG, 8, NHEAD * HIDD, 0, stream);
    spatial_gat(bPROJ, sg, T_WIN, NHEAD, HIDD, bOUT);
    ew_kernel<<<cdiv((size_t)M * 1024, 256), 256, 0, stream>>>(bOUT, bOUT, nullptr, (size_t)M * 1024, 0);

    // ---- sg0: Fin=1024 -> 256 ----
    run_mlp3(bEW, 8, 1, 128, 128);
    g_gemm(bOUT, 1024, sg0.W, HIDD, nullptr, bPROJ, HIDD, M, 1024, HIDD, 0, stream);
    g_gemm(bEW, 8, sg0.We, HIDD, nullptr, bEPR, HIDD, EDG, 8, HIDD, 0, stream);
    spatial_gat(bPROJ, sg0, T_WIN, NHEAD, HIDD / NHEAD, bACT1);
    ew_kernel<<<cdiv((size_t)M * HIDD, 256), 256, 0, stream>>>(bACT1, bACT1, nullptr, (size_t)M * HIDD, 0);

    // ---- sg1: 256 -> 256 ----
    run_mlp3(bEW, 8, 2, 128, 128);
    g_gemm(bACT1, HIDD, sg1.W, HIDD, nullptr, bPROJ, HIDD, M, HIDD, HIDD, 0, stream);
    g_gemm(bEW, 8, sg1.We, HIDD, nullptr, bEPR, HIDD, EDG, 8, HIDD, 0, stream);
    spatial_gat(bPROJ, sg1, T_WIN, NHEAD, HIDD / NHEAD, bACT2);
    ew_kernel<<<cdiv((size_t)M * HIDD, 256), 256, 0, stream>>>(bACT2, bACT2, nullptr, (size_t)M * HIDD, 0);

    // ---- bn_nodes (sn) ----
    bn_nodes_stats_kernel<<<NN, 256, 0, stream>>>(bACT2, bMU, bVAR, T_WIN, NN, HIDD);
    bn_nodes_apply_kernel<<<cdiv((size_t)M * HIDD, 256), 256, 0, stream>>>(bACT2, bMU, bVAR, sn_g, sn_b, T_WIN, NN, HIDD);

    // ---- temporal GAT chain (att == 1 exactly) ----
    float* tA = bACT1; float* tB = bOUT; const float* tin = bACT2;
    for (int l = 0; l < 4; ++l) {
        g_gemm(tin, HIDD, tg1[l].W, HIDD, nullptr, bPROJ, HIDD, M, HIDD, HIDD, 0, stream);
        float* tout = (l & 1) ? tB : tA;
        temporal_kernel<<<cdiv((size_t)M * HIDD, 256), 256, 0, stream>>>(bPROJ, tg1[l].b, tout, NN, HIDD, M * HIDD);
        tin = tout;
    }
    // tg = bn_feat(tg + t_), t_ in bOUT, tg in bACT2 -> result bACT1
    bn_feat_stats_kernel<<<HIDD, 256, 0, stream>>>(bACT2, bOUT, bMU, bVAR, M, HIDD);
    bn_feat_apply_kernel<<<cdiv((size_t)M * HIDD, 256), 256, 0, stream>>>(bACT2, bOUT, bACT1, bMU, bVAR, n1_g, n1_b, M, HIDD);

    // ---- diff_pool1 ----
    g_gemm(bACT1, HIDD, pool1_W, KK1, nullptr, bS, KK1, M, HIDD, KK1, 0, stream);
    softmax_rows_kernel<<<cdiv(M, 256), 256, 0, stream>>>(bS, M, KK1);
    pool_x_kernel<<<cdiv((size_t)KK1 * NN * HIDD, 256), 256, 0, stream>>>(bS, bACT1, bXP, T_WIN, NN, HIDD, KK1);
    pool_A_kernel<<<cdiv((size_t)NN * KK1 * KK1, 256), 256, 0, stream>>>(bS, bAP, T_WIN, NN, KK1);

    const int M1 = KK1 * NN;   // 7168
    // ---- sg2 (residual) ----
    run_mlp3(bEW, 8, 3, 32, 128);
    g_gemm(bXP, HIDD, sg2g.W, HIDD, nullptr, bPROJ, HIDD, M1, HIDD, HIDD, 0, stream);
    g_gemm(bEW, 8, sg2g.We, HIDD, nullptr, bEPR, HIDD, EDG, 8, HIDD, 0, stream);
    spatial_gat(bPROJ, sg2g, KK1, NHEAD, HIDD / NHEAD, bOUT);
    ew_kernel<<<cdiv((size_t)M1 * HIDD, 256), 256, 0, stream>>>(bACT2, bXP, bOUT, (size_t)M1 * HIDD, 1);

    // ---- tg2 (dense per-(n,l) attention, residual) ----
    g_gemm(bACT2, HIDD, tg2g.W, HIDD, nullptr, bPROJ, HIDD, M1, HIDD, HIDD, 0, stream);
    dot_heads_kernel<<<cdiv((size_t)M1 * NHEAD, 256), 256, 0, stream>>>(bPROJ, tg2g.asrc, bHSD, M1, NHEAD, HIDD / NHEAD);
    dot_heads_kernel<<<cdiv((size_t)M1 * NHEAD, 256), 256, 0, stream>>>(bPROJ, tg2g.adst, bHDD, M1, NHEAD, HIDD / NHEAD);
    tg2_kernel<<<NN * KK1, 64, 0, stream>>>(bPROJ, bHSD, bHDD, bAP, tg2g.We, tg2g.aedge,
                                            tg2g.b, bOUT, NN, KK1, NHEAD, HIDD / NHEAD);
    ew_kernel<<<cdiv((size_t)M1 * HIDD, 256), 256, 0, stream>>>(bACT1, bACT2, bOUT, (size_t)M1 * HIDD, 1);

    // ---- bn_feat (n2) ----
    bn_feat_stats_kernel<<<HIDD, 256, 0, stream>>>(bACT1, nullptr, bMU, bVAR, M1, HIDD);
    bn_feat_apply_kernel<<<cdiv((size_t)M1 * HIDD, 256), 256, 0, stream>>>(bACT1, nullptr, bACT1, bMU, bVAR, n2_g, n2_b, M1, HIDD);

    // ---- diff_pool2 (only x2 is consumed downstream) ----
    g_gemm(bACT1, HIDD, pool2_W, KK2, nullptr, bS, KK2, M1, HIDD, KK2, 0, stream);
    softmax_rows_kernel<<<cdiv(M1, 256), 256, 0, stream>>>(bS, M1, KK2);
    pool_x_kernel<<<cdiv((size_t)KK2 * NN * HIDD, 256), 256, 0, stream>>>(bS, bACT1, bX2, KK1, NN, HIDD, KK2);

    // ---- gout: H=1, C=26, output straight to d_out ----
    run_mlp3(bEW, 8, 4, 32, 128);
    const int M2 = KK2 * NN;   // 2560
    g_gemm(bX2, HIDD, goutp.W, FIN, nullptr, bPROJ, FIN, M2, HIDD, FIN, 0, stream);
    g_gemm(bEW, 8, goutp.We, FIN, nullptr, bEPR, FIN, EDG, 8, FIN, 0, stream);
    float* outp = (float*)d_out;
    dot_heads_kernel<<<cdiv((size_t)M2, 256), 256, 0, stream>>>(bPROJ, goutp.asrc, bHSD, M2, 1, FIN);
    dot_heads_kernel<<<cdiv((size_t)M2, 256), 256, 0, stream>>>(bPROJ, goutp.adst, bHDD, M2, 1, FIN);
    dot_heads_kernel<<<cdiv((size_t)EDG, 256), 256, 0, stream>>>(bEPR, goutp.aedge, bEDOT, EDG, 1, FIN);
    fill_kernel<<<cdiv((size_t)M2, 256), 256, 0, stream>>>(bMX, -3.0e38f, (size_t)M2);
    fill_kernel<<<cdiv((size_t)M2, 256), 256, 0, stream>>>(bDEN, 0.f, (size_t)M2);
    gat_logits_kernel<<<cdiv((size_t)KK2 * EDG, 256), 256, 0, stream>>>(bHSD, bHDD, bEDOT, e_src, e_dst, bEXV, bMX, KK2, NN, EDG, 1);
    gat_exp_den_kernel<<<cdiv((size_t)KK2 * EDG, 256), 256, 0, stream>>>(bEXV, bMX, bDEN, e_dst, KK2, NN, EDG, 1);
    bias_init_kernel<<<cdiv((size_t)M2 * FIN, 256), 256, 0, stream>>>(outp, goutp.b, (size_t)M2, FIN);
    gat_scatter_kernel<<<KK2 * EDG, 64, 0, stream>>>(bPROJ, bEXV, bDEN, e_src, e_dst, outp, KK2, NN, EDG, 1, FIN);
}
